// CIN_47236050321752
// MI455X (gfx1250) — matrix operand order repaired and checked by reading, not compiled, not run on hardware
//
#include <hip/hip_runtime.h>

// CIN (xDeepFM) fused 3-layer kernel for gfx1250 (MI455X).
// B=1024, N=D=32, layers H=128,128,128.
// Per-layer GEMM view over rows (b,d): out[(b,d),k] = Z @ W with
// Z[(b,d), h*32+n] = xk[b,h,d]*x0[b,n,d] generated in registers.
// ~77 GFLOP vs ~10 MB traffic -> compute bound -> v_wmma_f32_16x16x32_bf16
// with f32 accumulation; A-fragments built with v_pk_mul_bf16 (8 VALU/WMMA).

typedef __attribute__((ext_vector_type(16))) __bf16 v16bf;
typedef __attribute__((ext_vector_type(8)))  float  v8f;

union AFrag { unsigned int u[8]; v16bf v; };

// {bf16(f), bf16(f)} packed in one dword (single v_cvt_pk_bf16_f32)
__device__ inline unsigned int dupbf(float f) {
  unsigned int r;
  asm("v_cvt_pk_bf16_f32 %0, %1, %2" : "=v"(r) : "v"(f), "v"(f));
  return r;
}
__device__ inline unsigned int pkmul(unsigned int a, unsigned int b) {
  unsigned int r;
  asm("v_pk_mul_bf16 %0, %1, %2" : "=v"(r) : "v"(a), "v"(b));
  return r;
}

// ---------------------------------------------------------------------------
// Pack W_L (f32, (Kh*32,128)) into WMMA-B-fragment-ordered bf16 blocks.
// Block (h,ktile) = 32x16 tile; element order: lane=pos/16, e=pos%16,
// c=lane%16 (column), g=lane/16, kk=g*16+e (K-row). GEMM loads 32B/lane.
// ---------------------------------------------------------------------------
__global__ void cin_pack_w(const float* __restrict__ w,
                           __bf16* __restrict__ out, int total) {
  int idx = blockIdx.x * blockDim.x + threadIdx.x;
  if (idx >= total) return;
  int blk  = idx >> 9;
  int pos  = idx & 511;
  int h    = blk >> 3;
  int kt   = blk & 7;
  int lane = pos >> 4;
  int e    = pos & 15;
  int c    = lane & 15;
  int g    = lane >> 4;
  int kk   = g * 16 + e;
  out[idx] = (__bf16)w[(h * 32 + kk) * 128 + kt * 16 + c];
}

// ---------------------------------------------------------------------------
// Fused kernel: one WG (8 waves) per TWO batches. Wave wv owns k-columns
// [16wv,16wv+16) and both 16-row d-tiles, for both batches (4 accumulators),
// so each W fragment load feeds 4 WMMAs.
// LDS xk2 holds xk as duplicated-bf16 dwords: one ds_load_b32 = broadcast pair.
// ---------------------------------------------------------------------------
__global__ __launch_bounds__(256) void cin_kernel(const float* __restrict__ x0,
                                                  const __bf16* __restrict__ wb,
                                                  float* __restrict__ out) {
  __shared__ unsigned int xk2[2][128 * 32];   // [batch][h*32+d] = {v,v}, 32 KB

  const int tid  = threadIdx.x;
  const int lane = tid & 31;
  const int wv   = tid >> 5;      // wave id 0..7 -> k-tile
  const int m    = lane & 15;
  const int g    = lane >> 4;
  const int b0   = blockIdx.x * 2;

  // ---- Stage x0 for both batches (coalesced f32 -> dup-bf16 dwords) ----
#pragma unroll
  for (int p = 0; p < 2; ++p) {
    const float4* src = reinterpret_cast<const float4*>(x0 + (size_t)(b0 + p) * 1024);
    float4 v = src[tid];                     // 256 threads x 4 = 1024
    uint4 pk;
    pk.x = dupbf(v.x); pk.y = dupbf(v.y); pk.z = dupbf(v.z); pk.w = dupbf(v.w);
    *reinterpret_cast<uint4*>(&xk2[p][tid * 4]) = pk;
  }
  __syncthreads();

  // ---- Per-lane x0 column, packed bf16 pairs in A-fragment element order ----
  // A fragment in this code's k order (same as packed B): lane row M=m; element e -> n = g*16+e.
  // Dword j holds elements e=2j (lo), 2j+1 (hi); n(2j+1)=n(2j)+1.
  unsigned int x0pk[2][2][8];                // [batch][d-tile][j]
#pragma unroll
  for (int p = 0; p < 2; ++p)
#pragma unroll
    for (int t = 0; t < 2; ++t) {
      int d = t * 16 + m;
#pragma unroll
      for (int j = 0; j < 8; ++j) {
        int e  = 2 * j;
        int n0 = g * 16 + e;
        unsigned int lo = xk2[p][n0 * 32 + d];
        unsigned int hi = xk2[p][(n0 + 1) * 32 + d];
        x0pk[p][t][j] = (hi & 0xffff0000u) | (lo & 0x0000ffffu);
      }
    }

  for (int L = 0; L < 3; ++L) {
    const int Kh   = (L == 0) ? 32 : 128;
    const int base = (L == 0) ? 0 : ((L == 1) ? 32 * 8 * 512 : 160 * 8 * 512);
    const __bf16* wl = wb + base + (size_t)wv * 512 + (size_t)lane * 16;

    v8f z = {0.f, 0.f, 0.f, 0.f, 0.f, 0.f, 0.f, 0.f};
    v8f c[2][2];                              // [batch][d-tile]
#pragma unroll
    for (int p = 0; p < 2; ++p)
#pragma unroll
      for (int t = 0; t < 2; ++t) c[p][t] = z;

    // software-pipelined B-fragment load (contiguous 32 B/lane per K-block)
    v16bf bfr = *reinterpret_cast<const v16bf*>(wl);

#pragma unroll 2
    for (int h = 0; h < Kh; ++h) {
      int hn = (h + 1 < Kh) ? (h + 1) : h;
      v16bf bnx = *reinterpret_cast<const v16bf*>(wl + (size_t)hn * 4096);

      // broadcast xk scalars: dup-dword per (batch, d-tile)
      unsigned int xs[2][2];
      xs[0][0] = xk2[0][h * 32 + m];
      xs[0][1] = xk2[0][h * 32 + 16 + m];
      xs[1][0] = xk2[1][h * 32 + m];
      xs[1][1] = xk2[1][h * 32 + 16 + m];

      AFrag a[2][2];
#pragma unroll
      for (int j = 0; j < 8; ++j) {
        a[0][0].u[j] = pkmul(xs[0][0], x0pk[0][0][j]);
        a[0][1].u[j] = pkmul(xs[0][1], x0pk[0][1][j]);
        a[1][0].u[j] = pkmul(xs[1][0], x0pk[1][0][j]);
        a[1][1].u[j] = pkmul(xs[1][1], x0pk[1][1][j]);
      }
#pragma unroll
      for (int p = 0; p < 2; ++p)
#pragma unroll
        for (int t = 0; t < 2; ++t)
          c[p][t] = __builtin_amdgcn_wmma_f32_16x16x32_bf16(
              false, a[p][t].v, false, bfr, (short)0, c[p][t], false, false);

      bfr = bnx;
    }

    __syncthreads();   // all waves done reading xk2 for this layer

    // C/D layout: lane, VGPR r -> M(d) = r + 8g, N(k) = m.
    // Write xk_new as dup-bf16 dwords (2x ds_store_b128 per tile); pool in f32.
    const int k = wv * 16 + m;
#pragma unroll
    for (int p = 0; p < 2; ++p) {
      float s = 0.0f;
#pragma unroll
      for (int t = 0; t < 2; ++t) {
        uint4 lo, hi;
        lo.x = dupbf(c[p][t][0]); lo.y = dupbf(c[p][t][1]);
        lo.z = dupbf(c[p][t][2]); lo.w = dupbf(c[p][t][3]);
        hi.x = dupbf(c[p][t][4]); hi.y = dupbf(c[p][t][5]);
        hi.z = dupbf(c[p][t][6]); hi.w = dupbf(c[p][t][7]);
        int dbase = k * 32 + t * 16 + 8 * g;   // d = dbase + r, r=0..7
        *reinterpret_cast<uint4*>(&xk2[p][dbase])     = lo;
        *reinterpret_cast<uint4*>(&xk2[p][dbase + 4]) = hi;
#pragma unroll
        for (int r = 0; r < 8; ++r) s += c[p][t][r];
      }
      s += __shfl_xor(s, 16, 32);              // lanes l, l^16 share k
      if (g == 0) out[(size_t)(b0 + p) * 384 + L * 128 + k] = s;
    }
    __syncthreads();   // xk_new visible before next layer reads it
  }
}

extern "C" void kernel_launch(void* const* d_in, const int* in_sizes, int n_in,
                              void* d_out, int out_size, void* d_ws, size_t ws_size,
                              hipStream_t stream) {
  const float* x0 = (const float*)d_in[0];
  const float* w0 = (const float*)d_in[1];   // (1024, 128)
  const float* w1 = (const float*)d_in[2];   // (4096, 128)
  const float* w2 = (const float*)d_in[3];   // (4096, 128)
  float* out = (float*)d_out;

  __bf16* wb = (__bf16*)d_ws;                // 2.25 MB packed bf16 W blocks
  const int T0 = 32 * 8 * 512;               // 131072
  const int T1 = 128 * 8 * 512;              // 524288
  const int T2 = 128 * 8 * 512;              // 524288

  cin_pack_w<<<(T0 + 255) / 256, 256, 0, stream>>>(w0, wb, T0);
  cin_pack_w<<<(T1 + 255) / 256, 256, 0, stream>>>(w1, wb + T0, T1);
  cin_pack_w<<<(T2 + 255) / 256, 256, 0, stream>>>(w2, wb + T0 + T1, T2);

  cin_kernel<<<512, 256, 0, stream>>>(x0, wb, out);
}
